// Informer_45500883534034
// MI455X (gfx1250) — matrix operand order repaired and checked by reading, not compiled, not run on hardware
//
#include <hip/hip_runtime.h>
#include <hip/hip_fp16.h>

typedef __attribute__((ext_vector_type(16))) _Float16 v16h;
typedef __attribute__((ext_vector_type(8)))  _Float16 v8h;
typedef __attribute__((ext_vector_type(8)))  float    v8f;

#define NBATCH 4096L
#define DM 512
#define DH 64
#define NHEADS 8
#define MAXL 40

__device__ __forceinline__ float gelu_exact(float x) {
  return 0.5f * x * (1.0f + erff(x * 0.7071067811865476f));
}

// ---------------------------------------------------------------------------
// WMMA GEMM:  C[M,N] = act( A[M,K] @ W[N,K]^T + bias (+ Res) )
// A is f16 row-major with leading dim `lda`; optional affine row remap for the
// wrap-padded conv input (rows_per_batch>0). Each wave computes a 32x64 tile:
// 2 A fragments x 4 B fragments -> 8x v_wmma_f32_16x16x32_f16 per 32-wide K
// step, with each B (weight) fragment reused across both M sub-tiles.
// ---------------------------------------------------------------------------
template<int ACT, bool HAS_RES, bool WRITE_F, bool WRITE_H>
__global__ void __launch_bounds__(128)
gemm_wmma(const _Float16* __restrict__ A, const _Float16* __restrict__ W,
          const float* __restrict__ bias, const float* __restrict__ Res,
          float* __restrict__ C, _Float16* __restrict__ Ch,
          int M, int N, int K, int lda, int rows_per_batch, int pad_rows)
{
  const int wave = blockIdx.x * 4 + (threadIdx.x >> 5);
  const int lane = threadIdx.x & 31;
  const int ntn  = N >> 6;                 // number of 64-wide N strips
  const int mt   = wave / ntn;             // 32-row M tiles
  const int nt   = wave - mt * ntn;
  if (mt * 32 >= M) return;

  // row remap (identity, or wrap-padded conv-window mapping)
  long arow[2];
  #pragma unroll
  for (int mi = 0; mi < 2; ++mi) {
    int am = mt * 32 + mi * 16 + (lane & 15);
    long r = am;
    if (rows_per_batch > 0) {
      int bb = am / rows_per_batch;
      r = (long)bb * (rows_per_batch + pad_rows) + (am - bb * rows_per_batch);
    }
    arow[mi] = r;
  }
  // A fragment: lanes 0-15 hold K=k..k+15; lanes 16-31 K+16.
  const _Float16* ap0 = A + arow[0] * (long)lda + ((lane & 16) ? 16 : 0);
  const _Float16* ap1 = A + arow[1] * (long)lda + ((lane & 16) ? 16 : 0);
  // B fragment: lane n holds W[n][k..k+15]; lanes 16-31 K+16.
  const _Float16* wp = W + ((long)(nt * 64) + (lane & 15)) * K + ((lane & 16) ? 16 : 0);

  v8f acc[2][4];
  #pragma unroll
  for (int mi = 0; mi < 2; ++mi)
    #pragma unroll
    for (int j = 0; j < 4; ++j)
      acc[mi][j] = (v8f){};

  for (int k = 0; k < K; k += 32) {
    v8h a0lo = *(const v8h*)(ap0 + k);
    v8h a0hi = *(const v8h*)(ap0 + k + 8);
    v8h a1lo = *(const v8h*)(ap1 + k);
    v8h a1hi = *(const v8h*)(ap1 + k + 8);
    v16h a0 = __builtin_shufflevector(a0lo, a0hi,
              0,1,2,3,4,5,6,7,8,9,10,11,12,13,14,15);
    v16h a1 = __builtin_shufflevector(a1lo, a1hi,
              0,1,2,3,4,5,6,7,8,9,10,11,12,13,14,15);
    v16h b0 = *(const v16h*)(wp + k);
    v16h b1 = *(const v16h*)(wp + (long)16 * K + k);
    v16h b2 = *(const v16h*)(wp + (long)32 * K + k);
    v16h b3 = *(const v16h*)(wp + (long)48 * K + k);
    acc[0][0] = __builtin_amdgcn_wmma_f32_16x16x32_f16(false, a0, false, b0, (short)0, acc[0][0], false, false);
    acc[1][0] = __builtin_amdgcn_wmma_f32_16x16x32_f16(false, a1, false, b0, (short)0, acc[1][0], false, false);
    acc[0][1] = __builtin_amdgcn_wmma_f32_16x16x32_f16(false, a0, false, b1, (short)0, acc[0][1], false, false);
    acc[1][1] = __builtin_amdgcn_wmma_f32_16x16x32_f16(false, a1, false, b1, (short)0, acc[1][1], false, false);
    acc[0][2] = __builtin_amdgcn_wmma_f32_16x16x32_f16(false, a0, false, b2, (short)0, acc[0][2], false, false);
    acc[1][2] = __builtin_amdgcn_wmma_f32_16x16x32_f16(false, a1, false, b2, (short)0, acc[1][2], false, false);
    acc[0][3] = __builtin_amdgcn_wmma_f32_16x16x32_f16(false, a0, false, b3, (short)0, acc[0][3], false, false);
    acc[1][3] = __builtin_amdgcn_wmma_f32_16x16x32_f16(false, a1, false, b3, (short)0, acc[1][3], false, false);
  }

  #pragma unroll
  for (int mi = 0; mi < 2; ++mi) {
    const int rbase = mt * 32 + mi * 16 + ((lane & 16) ? 8 : 0);
    #pragma unroll
    for (int j = 0; j < 4; ++j) {
      const int col = nt * 64 + j * 16 + (lane & 15);
      const float bv = bias ? bias[col] : 0.0f;
      #pragma unroll
      for (int r = 0; r < 8; ++r) {
        long idx = (long)(rbase + r) * N + col;
        float v = acc[mi][j][r] + bv;
        if (HAS_RES) v += Res[idx];
        if (ACT == 1) v = gelu_exact(v);
        if (WRITE_F) C[idx] = v;
        if (WRITE_H) Ch[idx] = (_Float16)v;
      }
    }
  }
}

// ------------------------- helper / elementwise kernels ---------------------
__global__ void f2h_kernel(const float* __restrict__ a, _Float16* __restrict__ o, long n) {
  for (long i = (long)blockIdx.x * blockDim.x + threadIdx.x; i < n;
       i += (long)gridDim.x * blockDim.x)
    o[i] = (_Float16)a[i];
}

// conv weight (O=512, I=512, KW=3) -> Wr[o, kw*512 + i] (f16), so that the
// padded-window GEMM with K=1536 computes the circular conv.
__global__ void conv_repack_kernel(const float* __restrict__ w, _Float16* __restrict__ Wr) {
  int idx = blockIdx.x * blockDim.x + threadIdx.x;
  if (idx >= 512 * 1536) return;
  int o  = idx / 1536;
  int kk = idx - o * 1536;
  int kw = kk >> 9;
  int i  = kk & 511;
  Wr[idx] = (_Float16)w[o * 1536 + i * 3 + kw];
}

// token embedding conv (ENC_IN=10, kw=3, circular) + sinusoidal pos-emb
__global__ void embed_kernel(const float* __restrict__ xe, const float* __restrict__ w,
                             float* __restrict__ X, _Float16* __restrict__ Xh, int L) {
  long idx = (long)blockIdx.x * blockDim.x + threadIdx.x;
  if (idx >= NBATCH * L * (long)DM) return;
  int d = idx & 511;
  long bl = idx >> 9;
  int l = (int)(bl % L);
  long b = bl / L;
  float acc = 0.f;
  for (int k = 0; k < 3; ++k) {
    int ls = l + k - 1; ls = (ls + L) % L;
    const float* xp = xe + ((long)b * L + ls) * 10;
    const float* wp = w + (long)d * 30 + k;       // w[d, i, k] at d*30 + i*3 + k
    for (int i = 0; i < 10; ++i) acc += wp[i * 3] * xp[i];
  }
  int half = d >> 1;
  float div = __expf((float)(2 * half) * (-9.210340371976184f / 512.f));
  float ang = (float)l * div;
  acc += (d & 1) ? __cosf(ang) : __sinf(ang);
  X[idx] = acc; Xh[idx] = (_Float16)acc;
}

// wrap-pad f16 activations along L: Xpad[b, 0]=x[b,L-1], 1..L = 0..L-1, L+1 = 0
__global__ void pad_wrap_kernel(const _Float16* __restrict__ Xh, _Float16* __restrict__ Xpad, int L) {
  long idx = (long)blockIdx.x * blockDim.x + threadIdx.x;
  if (idx >= NBATCH * (L + 2) * (long)DM) return;
  int d = idx & 511;
  long r = idx >> 9;
  int lp = (int)(r % (L + 2));
  long b = r / (L + 2);
  int ls = (lp - 1 + L) % L;
  Xpad[idx] = Xh[((long)b * L + ls) * DM + d];
}

// per-token LayerNorm over 512 channels; one wave (32 lanes x 16 vals) per row
__global__ void __launch_bounds__(128)
layernorm_kernel(const float* __restrict__ in, const float* __restrict__ g,
                 const float* __restrict__ be, float* __restrict__ out,
                 _Float16* __restrict__ outh, int M) {
  int wave = blockIdx.x * 4 + (threadIdx.x >> 5);
  int lane = threadIdx.x & 31;
  if (wave >= M) return;
  const float* row = in + (long)wave * DM;
  float v[16], s = 0.f;
  #pragma unroll
  for (int i = 0; i < 16; ++i) { v[i] = row[lane + i * 32]; s += v[i]; }
  #pragma unroll
  for (int o = 16; o > 0; o >>= 1) s += __shfl_xor(s, o, 32);
  float mean = s * (1.0f / 512.0f);
  float q = 0.f;
  #pragma unroll
  for (int i = 0; i < 16; ++i) { float d = v[i] - mean; q += d * d; }
  #pragma unroll
  for (int o = 16; o > 0; o >>= 1) q += __shfl_xor(q, o, 32);
  float inv = rsqrtf(q * (1.0f / 512.0f) + 1e-5f);
  #pragma unroll
  for (int i = 0; i < 16; ++i) {
    int c = lane + i * 32;
    float r = g[c] * (v[i] - mean) * inv + be[c];
    out[(long)wave * DM + c] = r;
    if (outh) outh[(long)wave * DM + c] = (_Float16)r;
  }
}

// ProbSparse attention for one (b,h): sampled M-scores -> top-u -> full softmax
// for selected queries; ctx = mean(V) elsewhere. 64 threads = Dh columns.
__global__ void __launch_bounds__(64)
prob_attn_kernel(const float* __restrict__ Qg, const float* __restrict__ Kg,
                 const float* __restrict__ Vg, _Float16* __restrict__ ctxh,
                 int Lq, int U, int layer) {
  const int bh = blockIdx.x;
  const int b = bh >> 3, h = bh & 7;
  const int t = threadIdx.x;                 // 0..63 (Dh)
  __shared__ float Qs[MAXL][65], Ks[MAXL][65], Vs[MAXL][65];
  __shared__ float Msc[MAXL], attn[MAXL];
  __shared__ int   top[MAXL];

  for (int l = 0; l < Lq; ++l) {
    long base = ((long)(b * Lq + l)) * DM + h * DH + t;
    Qs[l][t] = Qg[base]; Ks[l][t] = Kg[base]; Vs[l][t] = Vg[base];
  }
  __syncthreads();

  // sampled sparsity measure M[l] = max_s(q.k_s) - sum_s(q.k_s)/Lk
  for (int l = t; l < Lq; l += 64) {
    float mx = -1e30f, sm = 0.f;
    for (int s = 0; s < U; ++s) {
      unsigned hsh = (unsigned)(l * 9973 + s * 7919 + layer * 131071 + 12345);
      hsh ^= hsh >> 13; hsh *= 2654435761u; hsh ^= hsh >> 16;
      int j = (int)(hsh % (unsigned)Lq);
      float d = 0.f;
      for (int e = 0; e < DH; ++e) d += Qs[l][e] * Ks[j][e];
      mx = fmaxf(mx, d); sm += d;
    }
    Msc[l] = mx - sm / (float)Lq;
  }
  __syncthreads();

  if (t == 0) {                               // top-U selection (distinct)
    unsigned long long used = 0ull;
    for (int i = 0; i < U; ++i) {
      float best = -1e30f; int bj = 0;
      for (int j = 0; j < Lq; ++j) {
        if ((used >> j) & 1ull) continue;
        if (Msc[j] > best) { best = Msc[j]; bj = j; }
      }
      used |= 1ull << bj; top[i] = bj;
    }
  }
  __syncthreads();

  float vm = 0.f;
  for (int l = 0; l < Lq; ++l) vm += Vs[l][t];
  vm /= (float)Lq;
  for (int l = 0; l < Lq; ++l)
    ctxh[((long)(b * Lq + l)) * DM + h * DH + t] = (_Float16)vm;
  __syncthreads();

  for (int i = 0; i < U; ++i) {
    int lq = top[i];
    for (int j = t; j < Lq; j += 64) {
      float d = 0.f;
      for (int e = 0; e < DH; ++e) d += Qs[lq][e] * Ks[j][e];
      attn[j] = d * 0.125f;                   // 1/sqrt(64)
    }
    __syncthreads();
    if (t == 0) {
      float m = -1e30f;
      for (int j = 0; j < Lq; ++j) m = fmaxf(m, attn[j]);
      float s = 0.f;
      for (int j = 0; j < Lq; ++j) { attn[j] = __expf(attn[j] - m); s += attn[j]; }
      float inv = 1.f / s;
      for (int j = 0; j < Lq; ++j) attn[j] *= inv;
    }
    __syncthreads();
    float upd = 0.f;
    for (int j = 0; j < Lq; ++j) upd += attn[j] * Vs[j][t];
    ctxh[((long)(b * Lq + lq)) * DM + h * DH + t] = (_Float16)upd;
    __syncthreads();
  }
}

// batchnorm stats over (B,L) per channel (biased variance)
__global__ void __launch_bounds__(256)
bn_stats_kernel(const float* __restrict__ Y, float* __restrict__ mean,
                float* __restrict__ var, int Mtok) {
  int c = blockIdx.x;
  float s = 0.f, q = 0.f;
  for (long t = threadIdx.x; t < Mtok; t += 256) {
    float v = Y[t * DM + c]; s += v; q += v * v;
  }
  __shared__ float ss[256], qq[256];
  ss[threadIdx.x] = s; qq[threadIdx.x] = q; __syncthreads();
  for (int o = 128; o > 0; o >>= 1) {
    if (threadIdx.x < o) { ss[threadIdx.x] += ss[threadIdx.x + o]; qq[threadIdx.x] += qq[threadIdx.x + o]; }
    __syncthreads();
  }
  if (threadIdx.x == 0) {
    float m = ss[0] / (float)Mtok;
    mean[c] = m; var[c] = qq[0] / (float)Mtok - m * m;
  }
}

// BN apply + ELU + maxpool(k=3,s=2,-inf pad) fused; emits f32 + f16 next-x
__global__ void bn_elu_pool_kernel(const float* __restrict__ Y, const float* __restrict__ mean,
                                   const float* __restrict__ var, const float* __restrict__ g,
                                   const float* __restrict__ be, float* __restrict__ X,
                                   _Float16* __restrict__ Xh, int L, int Lout) {
  long idx = (long)blockIdx.x * blockDim.x + threadIdx.x;
  if (idx >= NBATCH * Lout * (long)DM) return;
  int d = idx & 511;
  long r = idx >> 9;
  int lo = (int)(r % Lout);
  long b = r / Lout;
  float m = mean[d], iv = rsqrtf(var[d] + 1e-5f), gg = g[d], bb = be[d];
  float best = -1e30f;
  for (int k = 0; k < 3; ++k) {
    int li = 2 * lo - 1 + k;
    if (li < 0 || li >= L) continue;
    float v = gg * (Y[((long)b * L + li) * DM + d] - m) * iv + bb;
    v = (v > 0.f) ? v : (__expf(v) - 1.f);
    best = fmaxf(best, v);
  }
  X[idx] = best; Xh[idx] = (_Float16)best;
}

// final projection 512 -> 1 per token (wave per token)
__global__ void __launch_bounds__(128)
proj_kernel(const float* __restrict__ X, const float* __restrict__ pw,
            const float* __restrict__ pb, float* __restrict__ y, int Mtok) {
  int wave = blockIdx.x * 4 + (threadIdx.x >> 5);
  int lane = threadIdx.x & 31;
  if (wave >= Mtok) return;
  const float* row = X + (long)wave * DM;
  float s = 0.f;
  for (int i = lane; i < DM; i += 32) s += row[i] * pw[i];
  #pragma unroll
  for (int o = 16; o > 0; o >>= 1) s += __shfl_xor(s, o, 32);
  if (lane == 0) y[wave] = s + pb[0];
}

// (B,10) @ (10,)  + b  -> (B,)
__global__ void proj2_kernel(const float* __restrict__ y, const float* __restrict__ w2,
                             const float* __restrict__ b2, float* __restrict__ out) {
  int b = blockIdx.x * blockDim.x + threadIdx.x;
  if (b >= (int)NBATCH) return;
  float s = 0.f;
  for (int l = 0; l < 10; ++l) s += y[b * 10 + l] * w2[l];
  out[b] = s + b2[0];
}

// ---------------------------------------------------------------------------
// Input order assumption: d_in[0] = x_enc, then params flattened with sorted
// dict keys (jax pytree order):
//   convs[0]{b,be,g,w} convs[1]{...} -> idx 1..8
//   emb_w -> 9
//   layers[i]{W1,W2,Wk,Wo,Wq,Wv,b1,b2,be1,be2,bk,bo,bq,bv,g1,g2} -> 10+i*16
//   norm_b 58, norm_g 59, proj2_b 60, proj2_w 61, proj_b 62, proj_w 63
// ---------------------------------------------------------------------------
extern "C" void kernel_launch(void* const* d_in, const int* in_sizes, int n_in,
                              void* d_out, int out_size, void* d_ws, size_t ws_size,
                              hipStream_t stream) {
  (void)in_sizes; (void)n_in; (void)out_size; (void)ws_size;
  const float* x_enc = (const float*)d_in[0];
  const float *conv_b[2], *conv_be[2], *conv_g[2], *conv_w[2];
  for (int c = 0; c < 2; ++c) {
    conv_b[c]  = (const float*)d_in[1 + 4 * c + 0];
    conv_be[c] = (const float*)d_in[1 + 4 * c + 1];
    conv_g[c]  = (const float*)d_in[1 + 4 * c + 2];
    conv_w[c]  = (const float*)d_in[1 + 4 * c + 3];
  }
  const float* emb_w = (const float*)d_in[9];
#define LPARM(i, w) ((const float*)d_in[10 + (i) * 16 + (w)])
  // W1=0 W2=1 Wk=2 Wo=3 Wq=4 Wv=5 b1=6 b2=7 be1=8 be2=9 bk=10 bo=11 bq=12 bv=13 g1=14 g2=15
  const float* norm_b  = (const float*)d_in[58];
  const float* norm_g  = (const float*)d_in[59];
  const float* proj2_b = (const float*)d_in[60];
  const float* proj2_w = (const float*)d_in[61];
  const float* proj_b  = (const float*)d_in[62];
  const float* proj_w  = (const float*)d_in[63];

  // workspace carve-up
  const long M0 = NBATCH * 40;
  const size_t SZF = (size_t)M0 * DM * 4;
  const size_t SZH = (size_t)M0 * DM * 2;
  char* p = (char*)d_ws;
  float*    X    = (float*)p;    p += SZF;  // current activations (f32)
  float*    Qf   = (float*)p;    p += SZF;  // Q / GEMM-out / conv-Y scratch
  float*    Kf   = (float*)p;    p += SZF;
  float*    Vf   = (float*)p;    p += SZF;
  _Float16* Xh   = (_Float16*)p; p += SZH;  // f16 mirror of x
  _Float16* Hb   = (_Float16*)p; p += SZH;  // ctx_h / ffn-hidden_h
  _Float16* Xpad = (_Float16*)p; p += (size_t)NBATCH * 42 * DM * 2;
  _Float16* Wh   = (_Float16*)p; p += (size_t)6291456 * 2;  // all f16 weights
  float* bn_mean = (float*)p;    p += 2048;
  float* bn_var  = (float*)p;    p += 2048;
  float* Ptmp    = (float*)p;    p += (size_t)NBATCH * 10 * 4;

  // convert dense weights to f16: per layer [Wq,Wk,Wv,Wo,W1,W2] @ 512*512
  const int widx[6] = {4, 2, 5, 3, 0, 1};
  for (int i = 0; i < 3; ++i)
    for (int w = 0; w < 6; ++w)
      f2h_kernel<<<1024, 256, 0, stream>>>(LPARM(i, widx[w]),
          Wh + (long)i * 6 * 262144 + (long)w * 262144, 262144);
  for (int c = 0; c < 2; ++c)
    conv_repack_kernel<<<(512 * 1536 + 255) / 256, 256, 0, stream>>>(
        conv_w[c], Wh + 4718592 + (long)c * 786432);

  // embedding + positional encoding
  embed_kernel<<<(int)((M0 * DM + 255) / 256), 256, 0, stream>>>(x_enc, emb_w, X, Xh, 40);

  int Lcur = 40;
  const int Uarr[3] = {20, 15, 10};   // min(5*ceil(ln L), L) for L=40,20,10
  for (int i = 0; i < 3; ++i) {
    const long M = NBATCH * Lcur;
    const int gblocks = (int)(((M / 32) * 8 + 3) / 4);   // waves = (M/32)*(512/64)
    const long wb = (long)i * 6 * 262144;

    gemm_wmma<0, false, true, false><<<gblocks, 128, 0, stream>>>(
        Xh, Wh + wb,              LPARM(i, 12), nullptr, Qf, nullptr, (int)M, 512, 512, 512, 0, 0);
    gemm_wmma<0, false, true, false><<<gblocks, 128, 0, stream>>>(
        Xh, Wh + wb + 262144,     LPARM(i, 10), nullptr, Kf, nullptr, (int)M, 512, 512, 512, 0, 0);
    gemm_wmma<0, false, true, false><<<gblocks, 128, 0, stream>>>(
        Xh, Wh + wb + 2 * 262144, LPARM(i, 13), nullptr, Vf, nullptr, (int)M, 512, 512, 512, 0, 0);

    prob_attn_kernel<<<(int)(NBATCH * NHEADS), 64, 0, stream>>>(Qf, Kf, Vf, Hb, Lcur, Uarr[i], i);

    gemm_wmma<0, true, true, false><<<gblocks, 128, 0, stream>>>(
        Hb, Wh + wb + 3 * 262144, LPARM(i, 11), X, Qf, nullptr, (int)M, 512, 512, 512, 0, 0);
    layernorm_kernel<<<(int)((M + 3) / 4), 128, 0, stream>>>(Qf, LPARM(i, 14), LPARM(i, 8), X, Xh, (int)M);

    gemm_wmma<1, false, false, true><<<gblocks, 128, 0, stream>>>(
        Xh, Wh + wb + 4 * 262144, LPARM(i, 6), nullptr, nullptr, Hb, (int)M, 512, 512, 512, 0, 0);
    gemm_wmma<0, true, true, false><<<gblocks, 128, 0, stream>>>(
        Hb, Wh + wb + 5 * 262144, LPARM(i, 7), X, Qf, nullptr, (int)M, 512, 512, 512, 0, 0);
    layernorm_kernel<<<(int)((M + 3) / 4), 128, 0, stream>>>(Qf, LPARM(i, 15), LPARM(i, 9), X, Xh, (int)M);

    if (i < 2) {
      const long npad = NBATCH * (Lcur + 2) * (long)DM;
      pad_wrap_kernel<<<(int)((npad + 255) / 256), 256, 0, stream>>>(Xh, Xpad, Lcur);
      gemm_wmma<0, false, true, false><<<gblocks, 128, 0, stream>>>(
          Xpad, Wh + 4718592 + (long)i * 786432, conv_b[i], nullptr, Qf, nullptr,
          (int)M, 512, 1536, 512, Lcur, 2);
      bn_stats_kernel<<<512, 256, 0, stream>>>(Qf, bn_mean, bn_var, (int)M);
      const int Lout = Lcur / 2;
      const long nout = NBATCH * Lout * (long)DM;
      bn_elu_pool_kernel<<<(int)((nout + 255) / 256), 256, 0, stream>>>(
          Qf, bn_mean, bn_var, conv_g[i], conv_be[i], X, Xh, Lcur, Lout);
      Lcur = Lout;
    }
  }

  const long Mf = NBATCH * Lcur;     // 40960 tokens
  layernorm_kernel<<<(int)((Mf + 3) / 4), 128, 0, stream>>>(X, norm_g, norm_b, X, Xh, (int)Mf);
  proj_kernel<<<(int)((Mf + 3) / 4), 128, 0, stream>>>(X, proj_w, proj_b, Ptmp, (int)Mf);
  proj2_kernel<<<(int)((NBATCH + 255) / 256), 256, 0, stream>>>(Ptmp, proj2_w, proj2_b, (float*)d_out);
}
